// CConformerBlock_12034498363844
// MI455X (gfx1250) — matrix operand order repaired and checked by reading, not compiled, not run on hardware
//
#include <hip/hip_runtime.h>
#include <hip/hip_bf16.h>
#include <math.h>

#define EPSV 1e-5f

typedef __attribute__((ext_vector_type(16))) _Float16 v16h;
typedef __attribute__((ext_vector_type(8)))  float    v8f;

union H16u { v16h v; uint4 q[2]; };

__device__ inline v16h lds_frag_a(const _Float16* p){
  H16u u; u.q[0] = *(const uint4*)p; u.q[1] = *(const uint4*)(p + 16); return u.v;
}
__device__ inline v16h lds_frag_b(const _Float16* p){
  H16u u; u.q[0] = *(const uint4*)p; u.q[1] = *(const uint4*)(p + 16); return u.v;
}
__device__ inline void st2h(_Float16* p, float a, float b){
  union { _Float16 h[2]; unsigned u; } t;
  t.h[0] = (_Float16)a; t.h[1] = (_Float16)b;
  *(unsigned*)p = t.u;
}
__device__ inline float sigm(float x){ return 1.f / (1.f + __expf(-x)); }

// ---------------------------------------------------------------------------
// Complex GEMM via WMMA f16.
//   accR = Ar*Br - Ai*Bi ; accI = Ar*Bi + Ai*Br
// TRANSB: 0 -> B[k][n] (ldb = row len), 1 -> B[n][k]
// BCPLX : B is complex-interleaved from single pointer Br (ldb in complex elems)
// AREAL : imaginary part of A forced to zero (attn probs @ v)
// MODE  : 0 plain, 1 attention scores (q @ k^T per bh), 2 attn @ v per bh
// epilogue: t = acc (+bias conv (br-bi, br+bi)) ; t = cswish(t) if ACT ;
//           t *= alpha ; t += res if RES ; store interleaved f32
// ---------------------------------------------------------------------------
template<int TRANSB,int BCPLX,int AREAL,int BIAS,int ACT,int RES,int MODE>
__global__ __launch_bounds__(256) void k_cgemm(
    const float* __restrict__ A, const float* __restrict__ Br, const float* __restrict__ Bi,
    const float* __restrict__ biasR, const float* __restrict__ biasI,
    const float* __restrict__ res, float* __restrict__ C,
    int M, int K, int N, int lda, int ldb, int ldc, float alpha, int bh0)
{
  __shared__ _Float16 lAr[128*40];
  __shared__ _Float16 lAi[128*40];
  __shared__ _Float16 lBr[64*40];
  __shared__ _Float16 lBi[64*40];
  __shared__ _Float16 lBn[64*40];

  const int tid = threadIdx.x;
  const int m0 = blockIdx.y * 128;
  const int n0 = blockIdx.x * 64;

  size_t aOff = 0, bOff = 0, cOff = 0;
  if (MODE == 1){
    int z = blockIdx.z, bh = bh0 + z, b = bh >> 3, h = bh & 7;
    aOff = ((size_t)b*512*lda + h*64)*2;
    bOff = ((size_t)b*512*ldb + h*64)*2;
    cOff = (size_t)z*512*512*2;
  } else if (MODE == 2){
    int z = blockIdx.z, bh = bh0 + z, b = bh >> 3, h = bh & 7;
    aOff = (size_t)z*512*512*2;
    bOff = ((size_t)b*512*ldb + 512 + h*64)*2;
    cOff = ((size_t)b*512*ldc + h*64)*2;
  }
  const float* Ab = A + aOff;
  const float* Bb = Br + bOff;     // BCPLX base (interleaved)
  float* Cb = C + cOff;

  const int wave = tid >> 5, lane = tid & 31;
  const int wm = wave >> 1, wn = wave & 1;
  const int l15 = lane & 15, lhi = lane >> 4;

  v8f accR[2][2], accI[2][2];
  for (int mi = 0; mi < 2; ++mi)
    for (int ni = 0; ni < 2; ++ni)
      for (int e = 0; e < 8; ++e){ accR[mi][ni][e] = 0.f; accI[mi][ni][e] = 0.f; }

  const int arow = tid >> 1;            // 0..127
  const int akh  = (tid & 1) * 16;      // complex-k offset 0/16

  for (int k0 = 0; k0 < K; k0 += 32){
    __syncthreads();
    // ---- stage A tile (128 x 32 complex) -> f16 LDS, row stride 40 halfs
    {
      const float4* ap = (const float4*)(Ab + ((size_t)(m0 + arow)*lda + (k0 + akh))*2);
      #pragma unroll
      for (int kk = 0; kk < 8; ++kk){
        float4 v = ap[kk];
        int k = akh + kk*2;
        st2h(&lAr[arow*40 + k], v.x, v.z);
        if (AREAL) st2h(&lAi[arow*40 + k], 0.f, 0.f);
        else       st2h(&lAi[arow*40 + k], v.y, v.w);
      }
    }
    // ---- stage B tile (32 x 64) -> f16 LDS col-major [n][k], plus -Bi
    if (TRANSB == 0){
      int n = tid & 63, kk0 = tid >> 6;
      #pragma unroll
      for (int p = 0; p < 8; ++p){
        int k = kk0 + p*4;
        float vr, vi;
        if (BCPLX){ float2 t = *(const float2*)(Bb + ((size_t)(k0+k)*ldb + (n0+n))*2); vr = t.x; vi = t.y; }
        else { vr = Br[(size_t)(k0+k)*ldb + (n0+n)]; vi = Bi[(size_t)(k0+k)*ldb + (n0+n)]; }
        lBr[n*40 + k] = (_Float16)vr;
        lBi[n*40 + k] = (_Float16)vi;
        lBn[n*40 + k] = (_Float16)(-vi);
      }
    } else {
      int k = tid & 31, nn = tid >> 5;
      #pragma unroll
      for (int p = 0; p < 8; ++p){
        int n = nn + p*8;
        float vr, vi;
        if (BCPLX){ float2 t = *(const float2*)(Bb + ((size_t)(n0+n)*ldb + (k0+k))*2); vr = t.x; vi = t.y; }
        else { vr = Br[(size_t)(n0+n)*ldb + (k0+k)]; vi = Bi[(size_t)(n0+n)*ldb + (k0+k)]; }
        lBr[n*40 + k] = (_Float16)vr;
        lBi[n*40 + k] = (_Float16)vi;
        lBn[n*40 + k] = (_Float16)(-vi);
      }
    }
    __syncthreads();

    // ---- fragments per ISA layouts
    v16h fAr[2], fAi[2], fBr[2], fBi[2], fBn[2];
    #pragma unroll
    for (int mi = 0; mi < 2; ++mi){
      int base = (wm*32 + mi*16 + l15)*40 + lhi*8;
      fAr[mi] = lds_frag_a(&lAr[base]);
      if (!AREAL) fAi[mi] = lds_frag_a(&lAi[base]);
    }
    #pragma unroll
    for (int ni = 0; ni < 2; ++ni){
      int base = (wn*32 + ni*16 + l15)*40 + lhi*8;
      fBr[ni] = lds_frag_b(&lBr[base]);
      fBi[ni] = lds_frag_b(&lBi[base]);
      if (!AREAL) fBn[ni] = lds_frag_b(&lBn[base]);
    }
    #pragma unroll
    for (int mi = 0; mi < 2; ++mi)
      #pragma unroll
      for (int ni = 0; ni < 2; ++ni){
        accR[mi][ni] = __builtin_amdgcn_wmma_f32_16x16x32_f16(
            false, fAr[mi], false, fBr[ni], (short)0, accR[mi][ni], false, false);
        accI[mi][ni] = __builtin_amdgcn_wmma_f32_16x16x32_f16(
            false, fAr[mi], false, fBi[ni], (short)0, accI[mi][ni], false, false);
        if (!AREAL){
          accR[mi][ni] = __builtin_amdgcn_wmma_f32_16x16x32_f16(
              false, fAi[mi], false, fBn[ni], (short)0, accR[mi][ni], false, false);
          accI[mi][ni] = __builtin_amdgcn_wmma_f32_16x16x32_f16(
              false, fAi[mi], false, fBr[ni], (short)0, accI[mi][ni], false, false);
        }
      }
  }

  // ---- epilogue
  #pragma unroll
  for (int mi = 0; mi < 2; ++mi)
    #pragma unroll
    for (int ni = 0; ni < 2; ++ni){
      int col = n0 + wn*32 + ni*16 + l15;
      float bR = 0.f, bI = 0.f;
      if (BIAS){ float pr = biasR[col], pi = biasI[col]; bR = pr - pi; bI = pr + pi; }
      #pragma unroll
      for (int v = 0; v < 8; ++v){
        int row = m0 + wm*32 + mi*16 + lhi*8 + v;
        float r = accR[mi][ni][v], q = accI[mi][ni][v];
        if (BIAS){ r += bR; q += bI; }
        if (ACT){
          float sr = sigm(r), sq = sigm(q);
          float nr = r*sr - q*sq, nq = r*sq + q*sr;
          r = nr; q = nq;
        }
        r *= alpha; q *= alpha;
        size_t off = ((size_t)row*ldc + col)*2;
        if (RES){ float2 t = *(const float2*)(res + off); r += t.x; q += t.y; }
        float2 o; o.x = r; o.y = q;
        *(float2*)(Cb + off) = o;
      }
    }
  (void)M; (void)N;
}

// ---------------------------------------------------------------------------
// Complex layernorm: one block per row of 512 complex elements
// ---------------------------------------------------------------------------
__global__ __launch_bounds__(256) void k_cln(
    const float* __restrict__ x, const float* __restrict__ g_r, const float* __restrict__ b_r,
    const float* __restrict__ g_i, const float* __restrict__ b_i, float* __restrict__ y)
{
  __shared__ float4 red[256];
  const int row = blockIdx.x, tid = threadIdx.x;
  const float* xr = x + (size_t)row*1024;
  float2 v0 = *(const float2*)(xr + tid*2);
  float2 v1 = *(const float2*)(xr + (tid + 256)*2);
  float4 acc;
  acc.x = v0.x + v1.x;
  acc.y = v0.x*v0.x + v1.x*v1.x;
  acc.z = v0.y + v1.y;
  acc.w = v0.y*v0.y + v1.y*v1.y;
  red[tid] = acc; __syncthreads();
  for (int s = 128; s > 0; s >>= 1){
    if (tid < s){
      red[tid].x += red[tid+s].x; red[tid].y += red[tid+s].y;
      red[tid].z += red[tid+s].z; red[tid].w += red[tid+s].w;
    }
    __syncthreads();
  }
  const float inv512 = 1.f/512.f;
  float mr = red[0].x*inv512, vr = red[0].y*inv512 - mr*mr;
  float mi = red[0].z*inv512, vi = red[0].w*inv512 - mi*mi;
  float sr = rsqrtf(vr + EPSV), si = rsqrtf(vi + EPSV);
  float* yr = y + (size_t)row*1024;
  float2 o0, o1;
  o0.x = (v0.x - mr)*sr*g_r[tid] + b_r[tid];
  o0.y = (v0.y - mi)*si*g_i[tid] + b_i[tid];
  o1.x = (v1.x - mr)*sr*g_r[tid+256] + b_r[tid+256];
  o1.y = (v1.y - mi)*si*g_i[tid+256] + b_i[tid+256];
  *(float2*)(yr + tid*2) = o0;
  *(float2*)(yr + (tid+256)*2) = o1;
}

// ---------------------------------------------------------------------------
// Softmax over j with relative-position term added; writes probs into the
// real slot of S in place. One block per (i, z). q is pre-scaled by 1/8.
// ---------------------------------------------------------------------------
__global__ __launch_bounds__(256) void k_softmax_rpe(
    float* __restrict__ S, const float* __restrict__ Q,
    const float* __restrict__ rpe, int bh0)
{
  __shared__ float qre[64], qim[64], red[256];
  const int i = blockIdx.x, z = blockIdx.y, tid = threadIdx.x;
  const int bh = bh0 + z, b = bh >> 3, h = bh & 7;
  if (tid < 64){
    float2 t = *(const float2*)(Q + (((size_t)(b*512 + i))*512 + h*64 + tid)*2);
    qre[tid] = t.x; qim[tid] = t.y;
  }
  __syncthreads();
  float* Sr = S + ((size_t)z*512 + i)*512*2;
  float mval[2];
  #pragma unroll
  for (int rep = 0; rep < 2; ++rep){
    int j = tid + rep*256;
    float dr = Sr[j*2], di = Sr[j*2 + 1];
    const float* rp = rpe + (size_t)(i - j + 512)*64;
    float ar = 0.f, ai = 0.f;
    #pragma unroll 8
    for (int d = 0; d < 64; ++d){ float w = rp[d]; ar += qre[d]*w; ai += qim[d]*w; }
    dr += ar; di += ai;
    mval[rep] = sqrtf(dr*dr + di*di);
  }
  red[tid] = fmaxf(mval[0], mval[1]); __syncthreads();
  for (int s = 128; s > 0; s >>= 1){ if (tid < s) red[tid] = fmaxf(red[tid], red[tid+s]); __syncthreads(); }
  float mx = red[0];
  __syncthreads();
  float e0 = __expf(mval[0] - mx), e1 = __expf(mval[1] - mx);
  red[tid] = e0 + e1; __syncthreads();
  for (int s = 128; s > 0; s >>= 1){ if (tid < s) red[tid] += red[tid+s]; __syncthreads(); }
  float inv = 1.f / red[0];
  Sr[tid*2]       = e0*inv;
  Sr[(tid+256)*2] = e1*inv;
}

// ---------------------------------------------------------------------------
// GLU for conv module: H (M,2048,2) -> G (M,1024,2)
// ---------------------------------------------------------------------------
__global__ __launch_bounds__(256) void k_glu(const float* __restrict__ H, float* __restrict__ G)
{
  size_t idx = (size_t)blockIdx.x*256 + threadIdx.x;   // < 4096*1024
  size_t m = idx >> 10; int c = (int)(idx & 1023);
  float2 u = *(const float2*)(H + ((m*2048) + c)*2);
  float2 g = *(const float2*)(H + ((m*2048) + 1024 + c)*2);
  float gr = sigm(g.x), gi = sigm(g.y);
  float2 o; o.x = u.x*gr - u.y*gi; o.y = u.x*gi + u.y*gr;
  *(float2*)(G + idx*2) = o;
}

// ---------------------------------------------------------------------------
// Depthwise complex conv, K=31, same padding; layout (b*512+n, c, 2)
// ---------------------------------------------------------------------------
__global__ __launch_bounds__(256) void k_dwconv(
    const float* __restrict__ G, const float* __restrict__ Wdr, const float* __restrict__ Wdi,
    const float* __restrict__ bdr, const float* __restrict__ bdi, float* __restrict__ D)
{
  __shared__ float2 sv[158];
  const int b = blockIdx.z, c = blockIdx.y, n0 = blockIdx.x*128;
  for (int li = threadIdx.x; li < 158; li += 256){
    int n = n0 - 15 + li;
    float2 v; v.x = 0.f; v.y = 0.f;
    if (n >= 0 && n < 512) v = *(const float2*)(G + (((size_t)(b*512 + n))*1024 + c)*2);
    sv[li] = v;
  }
  __syncthreads();
  int t = threadIdx.x;
  if (t < 128){
    float cr = 0.f, ci = 0.f;
    #pragma unroll
    for (int tap = 0; tap < 31; ++tap){
      float wr = Wdr[c*31 + tap], wi = Wdi[c*31 + tap];
      float2 v = sv[t + tap];
      cr += v.x*wr - v.y*wi;
      ci += v.y*wr + v.x*wi;
    }
    cr += bdr[c] - bdi[c];
    ci += bdr[c] + bdi[c];
    float2 o; o.x = cr; o.y = ci;
    *(float2*)(D + (((size_t)(b*512 + n0 + t))*1024 + c)*2) = o;
  }
}

// ---------------------------------------------------------------------------
// BatchNorm stats per channel over 4096 (b,n) samples
// ---------------------------------------------------------------------------
__global__ __launch_bounds__(256) void k_bn_stats(const float* __restrict__ D, float* __restrict__ stats)
{
  __shared__ float4 red[256];
  const int c = blockIdx.x, tid = threadIdx.x;
  float sr = 0.f, sr2 = 0.f, si = 0.f, si2 = 0.f;
  for (int m = tid; m < 4096; m += 256){
    float2 t = *(const float2*)(D + ((size_t)m*1024 + c)*2);
    sr += t.x; sr2 += t.x*t.x; si += t.y; si2 += t.y*t.y;
  }
  float4 a; a.x = sr; a.y = sr2; a.z = si; a.w = si2;
  red[tid] = a; __syncthreads();
  for (int s = 128; s > 0; s >>= 1){
    if (tid < s){
      red[tid].x += red[tid+s].x; red[tid].y += red[tid+s].y;
      red[tid].z += red[tid+s].z; red[tid].w += red[tid+s].w;
    }
    __syncthreads();
  }
  if (tid == 0){
    const float inv = 1.f/4096.f;
    float mr = red[0].x*inv, vr = red[0].y*inv - mr*mr;
    float mi = red[0].z*inv, vi = red[0].w*inv - mi*mi;
    stats[c*4 + 0] = mr; stats[c*4 + 1] = rsqrtf(vr + EPSV);
    stats[c*4 + 2] = mi; stats[c*4 + 3] = rsqrtf(vi + EPSV);
  }
}

// ---------------------------------------------------------------------------
// BN apply + complex swish
// ---------------------------------------------------------------------------
__global__ __launch_bounds__(256) void k_bn_swish(
    const float* __restrict__ D, const float* __restrict__ stats,
    const float* __restrict__ gr, const float* __restrict__ br,
    const float* __restrict__ gi, const float* __restrict__ bi, float* __restrict__ G)
{
  size_t idx = (size_t)blockIdx.x*256 + threadIdx.x;  // < 4096*1024
  int c = (int)(idx & 1023);
  float2 t = *(const float2*)(D + idx*2);
  float a = (t.x - stats[c*4 + 0])*stats[c*4 + 1]*gr[c] + br[c];
  float b = (t.y - stats[c*4 + 2])*stats[c*4 + 3]*gi[c] + bi[c];
  float sa = sigm(a), sb = sigm(b);
  float2 o; o.x = a*sa - b*sb; o.y = a*sb + b*sa;
  *(float2*)(G + idx*2) = o;
}

// ---------------------------------------------------------------------------
// Host side
// ---------------------------------------------------------------------------
template<int TRANSB,int BCPLX,int AREAL,int BIAS,int ACT,int RES,int MODE>
static void cgemm(hipStream_t st, const float* A, const float* Br, const float* Bi,
                  const float* bR, const float* bI, const float* res, float* C,
                  int M, int K, int N, int lda, int ldb, int ldc, float alpha,
                  int zcnt, int bh0)
{
  dim3 g(N/64, M/128, zcnt);
  k_cgemm<TRANSB,BCPLX,AREAL,BIAS,ACT,RES,MODE><<<g, dim3(256), 0, st>>>(
      A, Br, Bi, bR, bI, res, C, M, K, N, lda, ldb, ldc, alpha, bh0);
}

// logical -> jax-sorted-pytree position (params sorted, x last)
static const int SORTED[62] = {
  61,
  44,42,43,41,            // ff1.ln gr,br,gi,bi
  34,33,38,37,36,35,40,39,// ff1 W1r,W1i,b1r,b1i,W2r,W2i,b2r,b2i
  56,54,55,53,            // ff2.ln
  46,45,50,49,48,47,52,51,// ff2 weights
  11,9,10,8,              // attn.ln
  5,4,1,0,3,2,7,6,12,     // Wq_r,Wq_i,Wkv_r,Wkv_i,Wo_r,Wo_i,bo_r,bo_i,rpe
  32,30,31,29,            // conv.ln
  14,13,20,19,            // cW1r,cW1i,cb1r,cb1i
  18,17,24,23,            // Wd_r,Wd_i,bd_r,bd_i
  28,26,27,25,            // bng_r,bnb_r,bng_i,bnb_i
  16,15,22,21,            // cW2r,cW2i,cb2r,cb2i
  60,58,59,57             // post gr,br,gi,bi
};

extern "C" void kernel_launch(void* const* d_in, const int* in_sizes, int n_in,
                              void* d_out, int out_size, void* d_ws, size_t ws_size,
                              hipStream_t stream) {
  // ----- resolve flatten order -----
  const bool xFirst = (n_in > 0 && in_sizes[0] == 8*512*512*2);
  const bool insertion = xFirst && (n_in > 1 && in_sizes[1] == 512);
  const bool xFirstSorted = xFirst && !insertion;
  auto IN = [&](int l)->const float* {
    int p;
    if (insertion)          p = l;
    else if (xFirstSorted)  p = (l == 0) ? 0 : SORTED[l] + 1;
    else                    p = SORTED[l];
    return (const float*)d_in[p];
  };

  // ----- scratch layout -----
  float* W = (float*)d_ws;
  size_t off = 0;
  auto take = [&](size_t n)->float* {
    float* p = W + off;
    off += (n + 255) & ~(size_t)255;
    return p;
  };
  float* A    = take((size_t)4096*512*2);    // cln outputs
  float* Hh   = take((size_t)4096*2048*2);   // FF hidden / conv pw1 out
  float* X1   = take((size_t)4096*512*2);
  float* X2   = take((size_t)4096*512*2);
  float* Qb   = take((size_t)4096*512*2);
  float* KVb  = take((size_t)4096*1024*2);
  float* AOb  = take((size_t)4096*512*2);
  float* Sb   = take((size_t)16*512*512*2);  // attention score chunk (16 bh)
  float* Gb   = take((size_t)4096*1024*2);
  float* Db   = take((size_t)4096*1024*2);
  float* STAT = take((size_t)1024*4);
  (void)ws_size; (void)out_size;

  const float* x0 = IN(0);
  float* out = (float*)d_out;

  // ===== FF1: x1 = 0.5*ff(cln(x)) + x =====
  k_cln<<<4096, 256, 0, stream>>>(x0, IN(1), IN(2), IN(3), IN(4), A);
  cgemm<0,0,0,1,1,0,0>(stream, A, IN(5), IN(6), IN(7), IN(8), nullptr, Hh,
                       4096, 512, 2048, 512, 2048, 2048, 1.0f, 1, 0);
  cgemm<0,0,0,1,0,1,0>(stream, Hh, IN(9), IN(10), IN(11), IN(12), x0, X1,
                       4096, 2048, 512, 2048, 512, 512, 0.5f, 1, 0);

  // ===== Attention: x2 = attn(cln(x1)) + x1 =====
  k_cln<<<4096, 256, 0, stream>>>(X1, IN(25), IN(26), IN(27), IN(28), A);
  cgemm<0,0,0,0,0,0,0>(stream, A, IN(29), IN(30), nullptr, nullptr, nullptr, Qb,
                       4096, 512, 512, 512, 512, 512, 0.125f, 1, 0);      // q, pre-scaled
  cgemm<0,0,0,0,0,0,0>(stream, A, IN(31), IN(32), nullptr, nullptr, nullptr, KVb,
                       4096, 512, 1024, 512, 1024, 1024, 1.0f, 1, 0);     // kv
  for (int bh0 = 0; bh0 < 64; bh0 += 16){
    cgemm<1,1,0,0,0,0,1>(stream, Qb, KVb, nullptr, nullptr, nullptr, nullptr, Sb,
                         512, 64, 512, 512, 1024, 512, 1.0f, 16, bh0);    // dr,di
    k_softmax_rpe<<<dim3(512,16), 256, 0, stream>>>(Sb, Qb, IN(37), bh0);
    cgemm<0,1,1,0,0,0,2>(stream, Sb, KVb, nullptr, nullptr, nullptr, nullptr, AOb,
                         512, 512, 64, 512, 1024, 512, 1.0f, 16, bh0);    // probs @ v
  }
  cgemm<0,0,0,1,0,1,0>(stream, AOb, IN(33), IN(34), IN(35), IN(36), X1, X2,
                       4096, 512, 512, 512, 512, 512, 1.0f, 1, 0);        // Wo + bo + res

  // ===== Conv module: x3 = conv(x2) + x2 =====
  k_cln<<<4096, 256, 0, stream>>>(X2, IN(38), IN(39), IN(40), IN(41), A);
  cgemm<1,0,0,1,0,0,0>(stream, A, IN(42), IN(43), IN(44), IN(45), nullptr, Hh,
                       4096, 512, 2048, 512, 512, 2048, 1.0f, 1, 0);      // pw1 (W stored (O,C))
  k_glu<<<16384, 256, 0, stream>>>(Hh, Gb);
  k_dwconv<<<dim3(4,1024,8), 256, 0, stream>>>(Gb, IN(46), IN(47), IN(48), IN(49), Db);
  k_bn_stats<<<1024, 256, 0, stream>>>(Db, STAT);
  k_bn_swish<<<16384, 256, 0, stream>>>(Db, STAT, IN(50), IN(51), IN(52), IN(53), Gb);
  cgemm<1,0,0,1,0,1,0>(stream, Gb, IN(54), IN(55), IN(56), IN(57), X2, X1,
                       4096, 1024, 512, 1024, 1024, 512, 1.0f, 1, 0);     // pw2 + res

  // ===== FF2: x4 = 0.5*ff(cln(x3)) + x3 =====
  k_cln<<<4096, 256, 0, stream>>>(X1, IN(13), IN(14), IN(15), IN(16), A);
  cgemm<0,0,0,1,1,0,0>(stream, A, IN(17), IN(18), IN(19), IN(20), nullptr, Hh,
                       4096, 512, 2048, 512, 2048, 2048, 1.0f, 1, 0);
  cgemm<0,0,0,1,0,1,0>(stream, Hh, IN(21), IN(22), IN(23), IN(24), X1, X2,
                       4096, 2048, 512, 2048, 512, 512, 0.5f, 1, 0);

  // ===== Post layernorm -> d_out =====
  k_cln<<<4096, 256, 0, stream>>>(X2, IN(58), IN(59), IN(60), IN(61), out);
}
